// FactorizedVectorQuantizer_51110110822812
// MI455X (gfx1250) — matrix operand order repaired and checked by reading, not compiled, not run on hardware
//
#include <hip/hip_runtime.h>

// ---------------------------------------------------------------------------
// FactorizedVectorQuantizer forward for MI455X (gfx1250, wave32, WMMA bf16)
// ---------------------------------------------------------------------------

typedef __bf16 bf16;
typedef __attribute__((ext_vector_type(16))) __bf16 v16bf;
typedef __attribute__((ext_vector_type(8)))  float  v8f;

// problem dims
constexpr int ROWS = 16 * 2048;     // B*T = 32768
constexpr int INCH = 512;
constexpr int LAT  = 256;
constexpr int CAT  = 768;           // 3*LAT
constexpr int VC = 1024, VP = 1024, VT = 512;
constexpr int VTOT = VC + VP + VT;  // 2560

// d_out layout (float elements), tuple order:
// x_recon [ROWS,512] | vq_loss | idx_c | idx_p | idx_t | zq_c | zq_p | zq_t
constexpr long O_RECON = 0;
constexpr long O_LOSS  = (long)ROWS * INCH;     // 16777216
constexpr long O_IDX   = O_LOSS + 1;
constexpr long O_ZQ    = O_IDX + 3L * ROWS;

// workspace layout (byte offsets, all 256B aligned)
constexpr size_t W_Z    = 0;                                // f32 Z[ROWS][768]
constexpr size_t W_CBBF = W_Z    + (size_t)ROWS * CAT * 4;  // bf16 cb[2560][256]
constexpr size_t W_CBN  = W_CBBF + (size_t)VTOT * LAT * 2;  // f32 ||cb||^2 [2560]
constexpr size_t W_IDX  = W_CBN  + (size_t)VTOT * 4;        // i32 idx[3][ROWS]
constexpr size_t W_LOSS = W_IDX  + (size_t)3 * ROWS * 4;    // f32 accumulator

#define DEV static __device__ __forceinline__

DEV bf16 f2bf(float f) {
  unsigned u = __builtin_bit_cast(unsigned, f);
  u += 0x7FFFu + ((u >> 16) & 1u);               // round-to-nearest-even
  unsigned short h = (unsigned short)(u >> 16);
  return __builtin_bit_cast(bf16, h);
}

// --- WMMA fragment loaders, per CDNA5 ISA 7.12.2 (wave32) -------------------
// 16-bit A matrix 16x32 (MxK): lanes 0-15 -> M=lane, K in {0..7,16..23};
//                              lanes 16-31 -> M=lane-16, K in {8..15,24..31}.
DEV v16bf load_frag_a_lds(const bf16* __restrict__ tile, int ldk, int kbase) {
  const int lane = threadIdx.x & 31;
  const int row  = lane & 15;
  const int kb   = kbase + ((lane & 16) ? 8 : 0);
  v16bf a{};
#pragma unroll
  for (int e = 0; e < 16; ++e) {
    const int k = kb + ((e >> 3) << 4) + (e & 7);
    a[e] = tile[row * ldk + k];
  }
  return a;
}

DEV v16bf load_frag_a_global(const bf16* __restrict__ base, int ld,
                             int vbase, int kbase) {
  const int lane = threadIdx.x & 31;
  const bf16* p = base + (size_t)(vbase + (lane & 15)) * ld;
  const int kb = kbase + ((lane & 16) ? 8 : 0);
  v16bf a{};
#pragma unroll
  for (int e = 0; e < 16; ++e) {
    const int k = kb + ((e >> 3) << 4) + (e & 7);
    a[e] = p[k];
  }
  return a;
}

// 16-bit B matrix 32x16 (KxN), stored in LDS as [n][k]:
// lanes 0-15 -> N=lane, K in {0..7,16..23}; lanes 16-31 -> N=lane-16, K in {8..15,24..31}.
DEV v16bf load_frag_b_lds(const bf16* __restrict__ tile, int ldk, int kbase) {
  const int lane = threadIdx.x & 31;
  const bf16* p = tile + (lane & 15) * ldk + kbase + ((lane & 16) ? 8 : 0);
  v16bf b{};
#pragma unroll
  for (int e = 0; e < 16; ++e) b[e] = p[((e >> 3) << 4) + (e & 7)];
  return b;
}

DEV v8f wmma_bf16(v16bf a, v16bf b, v8f c) {
  return __builtin_amdgcn_wmma_f32_16x16x32_bf16(
      /*neg_a=*/false, a, /*neg_b=*/false, b,
      /*c_mod=*/(short)0, c, /*reuse_a=*/false, /*reuse_b=*/false);
}

// ---------------------------------------------------------------------------
// Kernel: codebook prep (bf16 copy + row norms). One wave per codebook row.
// ---------------------------------------------------------------------------
__global__ __launch_bounds__(256)
void cbprep_kernel(const float* __restrict__ cbc, const float* __restrict__ cbp,
                   const float* __restrict__ cbt,
                   bf16* __restrict__ cbbf, float* __restrict__ cbn) {
  const int wave = threadIdx.x >> 5, lane = threadIdx.x & 31;
  const int row = blockIdx.x * 8 + wave;           // < 2560
  const float* src; int rrel;
  if (row < VC)           { src = cbc; rrel = row; }
  else if (row < VC + VP) { src = cbp; rrel = row - VC; }
  else                    { src = cbt; rrel = row - VC - VP; }
  float s = 0.f;
#pragma unroll
  for (int j = 0; j < 8; ++j) {
    const int d = lane + 32 * j;
    const float v = src[(size_t)rrel * LAT + d];
    cbbf[(size_t)row * LAT + d] = f2bf(v);
    s += v * v;
  }
#pragma unroll
  for (int off = 16; off > 0; off >>= 1) s += __shfl_xor(s, off, 32);
  if (lane == 0) cbn[row] = s;
}

// ---------------------------------------------------------------------------
// Kernel: projections  Z[ROWS,768] = X @ [Wc|Wp|Wt] + bias  (WMMA bf16)
// Block tile 64x64 (8 waves: 4 M-subtiles x 2 N-subtiles of 16x32).
// ---------------------------------------------------------------------------
__global__ __launch_bounds__(256)
void proj_kernel(const float* __restrict__ X,
                 const float* __restrict__ Wc, const float* __restrict__ bc,
                 const float* __restrict__ Wp, const float* __restrict__ bp,
                 const float* __restrict__ Wt, const float* __restrict__ bt,
                 float* __restrict__ Z) {
  constexpr int LDA = 40, LDB = 40;
  __shared__ bf16 At[64 * LDA];
  __shared__ bf16 Bt[64 * LDB];
  const int tid = threadIdx.x, lane = tid & 31, wave = tid >> 5;
  const int wm = wave & 3, wn = wave >> 2;
  const int m0 = blockIdx.x * 64;
  const int n0 = blockIdx.y * 64;
  const float* Wb; const float* bias;
  if (n0 < 256)      { Wb = Wc; bias = bc; }
  else if (n0 < 512) { Wb = Wp; bias = bp; }
  else               { Wb = Wt; bias = bt; }
  const int noff = n0 & 255;

  v8f acc0{}; v8f acc1{};
  for (int k0 = 0; k0 < INCH; k0 += 32) {
#pragma unroll
    for (int i = 0; i < 8; ++i) {                  // A: 64x32 fp32 -> bf16 LDS
      const int lin = tid + 256 * i;
      const int r = lin >> 5, k = lin & 31;
      At[r * LDA + k] = f2bf(X[(size_t)(m0 + r) * INCH + k0 + k]);
    }
#pragma unroll
    for (int i = 0; i < 8; ++i) {                  // B: 32x64 -> LDS [n][k]
      const int lin = tid + 256 * i;
      const int n = lin & 63, k = lin >> 6;
      Bt[n * LDB + k] = f2bf(Wb[(size_t)(k0 + k) * LAT + noff + n]);
    }
    __syncthreads();
    const v16bf a  = load_frag_a_lds(At + wm * 16 * LDA, LDA, 0);
    const v16bf b0 = load_frag_b_lds(Bt + (wn * 32) * LDB, LDB, 0);
    acc0 = wmma_bf16(a, b0, acc0);
    const v16bf b1 = load_frag_b_lds(Bt + (wn * 32 + 16) * LDB, LDB, 0);
    acc1 = wmma_bf16(a, b1, acc1);
    __syncthreads();
  }
  const int mrow = m0 + wm * 16 + ((lane & 16) ? 8 : 0);
  const int ncol = n0 + wn * 32 + (lane & 15);
  const int bi0  = noff + wn * 32 + (lane & 15);
#pragma unroll
  for (int r = 0; r < 8; ++r) {
    const int row = mrow + r;
    Z[(size_t)row * CAT + ncol]      = acc0[r] + bias[bi0];
    Z[(size_t)row * CAT + ncol + 16] = acc1[r] + bias[bi0 + 16];
  }
}

// ---------------------------------------------------------------------------
// Kernel: distance + argmin per branch.
// d(row,v) = ||cb_v||^2 - 2 * <z_row, cb_v>   (||z||^2 constant per row)
// Block: 16 z-rows in LDS as B; 8 waves sweep v-chunks (A from global bf16).
// ---------------------------------------------------------------------------
__global__ __launch_bounds__(256)
void dist_kernel(const float* __restrict__ Zb,   // branch column slice, stride CAT
                 const bf16* __restrict__ cb,    // [V][256] bf16
                 const float* __restrict__ cbn,  // [V] norms
                 int V, int* __restrict__ idxo) {
  __shared__ bf16 zt[16 * 256];
  __shared__ float rbest[8][16];
  __shared__ int   ridx[8][16];
  const int tid = threadIdx.x, lane = tid & 31, wave = tid >> 5;
  const int row0 = blockIdx.x * 16;
#pragma unroll
  for (int i = 0; i < 16; ++i) {
    const int lin = tid + 256 * i;
    const int r = lin >> 8, k = lin & 255;
    zt[lin] = f2bf(Zb[(size_t)(row0 + r) * CAT + k]);
  }
  __syncthreads();

  float best = 3.4e38f; int bestv = 0;
  const int nch = V >> 4;
  for (int m = wave; m < nch; m += 8) {
    if (m + 8 < nch)
      __builtin_prefetch(cb + (size_t)(m + 8) * 16 * LAT, 0, 1);
    v8f acc{};
#pragma unroll
    for (int kk = 0; kk < 8; ++kk) {
      const v16bf a = load_frag_a_global(cb, LAT, m * 16, kk * 32);
      const v16bf b = load_frag_b_lds(zt, 256, kk * 32);
      acc = wmma_bf16(a, b, acc);
    }
    const int vb = m * 16 + ((lane & 16) ? 8 : 0);
#pragma unroll
    for (int r2 = 0; r2 < 8; ++r2) {
      const int v = vb + r2;
      const float d = cbn[v] - 2.0f * acc[r2];
      if (d < best) { best = d; bestv = v; }
    }
  }
  // lanes l and l^16 hold the same z-row (N = lane&15): combine
  const float ob = __shfl_xor(best, 16, 32);
  const int   oi = __shfl_xor(bestv, 16, 32);
  if (ob < best || (ob == best && oi < bestv)) { best = ob; bestv = oi; }
  if (lane < 16) { rbest[wave][lane] = best; ridx[wave][lane] = bestv; }
  __syncthreads();
  if (tid < 16) {
    float bb = rbest[0][tid]; int bi = ridx[0][tid];
#pragma unroll
    for (int w = 1; w < 8; ++w) {
      const float c = rbest[w][tid]; const int ci = ridx[w][tid];
      if (c < bb || (c == bb && ci < bi)) { bb = c; bi = ci; }
    }
    idxo[row0 + tid] = bi;
  }
}

// ---------------------------------------------------------------------------
// Kernel: gather zq = cb[idx], emit idx (as float) and accumulate Sum (z-zq)^2
// One block per row; 256 threads = one latent element each, 3 branches.
// ---------------------------------------------------------------------------
__global__ __launch_bounds__(256)
void gather_kernel(const float* __restrict__ Z,
                   const float* __restrict__ cbc, const float* __restrict__ cbp,
                   const float* __restrict__ cbt,
                   const int* __restrict__ idx,
                   float* __restrict__ out, float* __restrict__ lossacc) {
  __shared__ float red[256];
  const int row = blockIdx.x;
  const int t = threadIdx.x;
  float local = 0.f;
#pragma unroll
  for (int br = 0; br < 3; ++br) {
    const float* cb = (br == 0) ? cbc : ((br == 1) ? cbp : cbt);
    const int id = idx[(size_t)br * ROWS + row];
    const float zq = cb[(size_t)id * LAT + t];
    out[O_ZQ + (size_t)br * ROWS * LAT + (size_t)row * LAT + t] = zq;
    const float z = Z[(size_t)row * CAT + br * LAT + t];
    const float df = z - zq;
    local += df * df;
    if (t == 0) out[O_IDX + (size_t)br * ROWS + row] = (float)id;
  }
  red[t] = local;
  __syncthreads();
  for (int s = 128; s > 0; s >>= 1) {
    if (t < s) red[t] += red[t + s];
    __syncthreads();
  }
  if (t == 0) atomicAdd(lossacc, red[0]);
}

// ---------------------------------------------------------------------------
// Kernel: reconstruction  x_recon[ROWS,512] = zq_concat[ROWS,768] @ Wr + br
// ---------------------------------------------------------------------------
__global__ __launch_bounds__(256)
void recon_kernel(const float* __restrict__ zq,   // out + O_ZQ (3 branch blocks)
                  const float* __restrict__ Wr, const float* __restrict__ br,
                  float* __restrict__ recon) {
  constexpr int LDA = 40, LDB = 40;
  __shared__ bf16 At[64 * LDA];
  __shared__ bf16 Bt[64 * LDB];
  const int tid = threadIdx.x, lane = tid & 31, wave = tid >> 5;
  const int wm = wave & 3, wn = wave >> 2;
  const int m0 = blockIdx.x * 64;
  const int n0 = blockIdx.y * 64;

  v8f acc0{}; v8f acc1{};
  for (int k0 = 0; k0 < CAT; k0 += 32) {
    const int brn = k0 >> 8;                       // whole 32-tile in one branch
    const float* src = zq + (size_t)brn * ROWS * LAT;
#pragma unroll
    for (int i = 0; i < 8; ++i) {                  // A: 64x32 from zq regions
      const int lin = tid + 256 * i;
      const int r = lin >> 5, k = lin & 31;
      At[r * LDA + k] =
          f2bf(src[(size_t)(m0 + r) * LAT + ((k0 & 255) + k)]);
    }
#pragma unroll
    for (int i = 0; i < 8; ++i) {                  // B: Wr[768,512] -> [n][k]
      const int lin = tid + 256 * i;
      const int n = lin & 63, k = lin >> 6;
      Bt[n * LDB + k] = f2bf(Wr[(size_t)(k0 + k) * INCH + n0 + n]);
    }
    __syncthreads();
    const v16bf a  = load_frag_a_lds(At + wm * 16 * LDA, LDA, 0);
    const v16bf b0 = load_frag_b_lds(Bt + (wn * 32) * LDB, LDB, 0);
    acc0 = wmma_bf16(a, b0, acc0);
    const v16bf b1 = load_frag_b_lds(Bt + (wn * 32 + 16) * LDB, LDB, 0);
    acc1 = wmma_bf16(a, b1, acc1);
    __syncthreads();
  }
  const int mrow = m0 + wm * 16 + ((lane & 16) ? 8 : 0);
  const int ncol = n0 + wn * 32 + (lane & 15);
#pragma unroll
  for (int r = 0; r < 8; ++r) {
    const int row = mrow + r;
    recon[(size_t)row * INCH + ncol]      = acc0[r] + br[ncol];
    recon[(size_t)row * INCH + ncol + 16] = acc1[r] + br[ncol + 16];
  }
}

__global__ void init_kernel(float* lossacc) {
  if (threadIdx.x == 0 && blockIdx.x == 0) *lossacc = 0.f;
}

__global__ void fin_kernel(const float* __restrict__ lossacc,
                           float* __restrict__ out) {
  if (threadIdx.x == 0 && blockIdx.x == 0) {
    // vq_loss = codebook + 0.25*commitment = 1.25/3 * mean over branches
    out[O_LOSS] = (*lossacc) * (1.25f / (3.0f * (float)ROWS * (float)LAT));
  }
}

// ---------------------------------------------------------------------------
extern "C" void kernel_launch(void* const* d_in, const int* in_sizes, int n_in,
                              void* d_out, int out_size, void* d_ws,
                              size_t ws_size, hipStream_t stream) {
  (void)in_sizes; (void)n_in; (void)out_size; (void)ws_size;
  const float* x   = (const float*)d_in[0];
  const float* Wc  = (const float*)d_in[1];
  const float* bc  = (const float*)d_in[2];
  const float* Wp  = (const float*)d_in[3];
  const float* bp  = (const float*)d_in[4];
  const float* Wt  = (const float*)d_in[5];
  const float* bt  = (const float*)d_in[6];
  const float* cbc = (const float*)d_in[7];
  const float* cbp = (const float*)d_in[8];
  const float* cbt = (const float*)d_in[9];
  const float* Wr  = (const float*)d_in[10];
  const float* brs = (const float*)d_in[11];
  float* out = (float*)d_out;
  char* ws = (char*)d_ws;

  float* Z    = (float*)(ws + W_Z);
  bf16*  cbbf = (bf16*)(ws + W_CBBF);
  float* cbn  = (float*)(ws + W_CBN);
  int*   idx  = (int*)(ws + W_IDX);
  float* lac  = (float*)(ws + W_LOSS);

  init_kernel<<<1, 32, 0, stream>>>(lac);
  cbprep_kernel<<<VTOT / 8, 256, 0, stream>>>(cbc, cbp, cbt, cbbf, cbn);
  proj_kernel<<<dim3(ROWS / 64, CAT / 64), 256, 0, stream>>>(
      x, Wc, bc, Wp, bp, Wt, bt, Z);
  dist_kernel<<<ROWS / 16, 256, 0, stream>>>(Z + 0, cbbf, cbn, VC, idx);
  dist_kernel<<<ROWS / 16, 256, 0, stream>>>(
      Z + 256, cbbf + (size_t)VC * LAT, cbn + VC, VP, idx + ROWS);
  dist_kernel<<<ROWS / 16, 256, 0, stream>>>(
      Z + 512, cbbf + (size_t)(VC + VP) * LAT, cbn + (VC + VP), VT,
      idx + 2 * ROWS);
  gather_kernel<<<ROWS, 256, 0, stream>>>(Z, cbc, cbp, cbt, idx, out, lac);
  recon_kernel<<<dim3(ROWS / 64, INCH / 64), 256, 0, stream>>>(
      out + O_ZQ, Wr, brs, out + O_RECON);
  fin_kernel<<<1, 32, 0, stream>>>(lac, out);
}
